// STGATPredictor_34187939676738
// MI455X (gfx1250) — matrix operand order repaired and checked by reading, not compiled, not run on hardware
//
#include <hip/hip_runtime.h>
#include <hip/hip_bf16.h>
#include <cstdint>

typedef _Float16 f16;
typedef __attribute__((ext_vector_type(8)))  _Float16 v8h;
typedef __attribute__((ext_vector_type(16))) _Float16 v16h;
typedef __attribute__((ext_vector_type(8)))  float    v8f;
typedef __attribute__((ext_vector_type(4)))  int      v4i_;

#define BN   6400   // B*N
#define NN   400
#define HB   16
#define TT   12
#define LL   3600   // edges
#define MP   448    // padded M for batched attention GEMM (400 -> 448)
#define KP   416    // padded K for batched attention GEMM (400 -> 416)
#define NPB  128    // padded B-rows for batched attention GEMM (32 -> 128)

#if defined(__has_builtin)
#  if __has_builtin(__builtin_amdgcn_global_load_async_to_lds_b128)
#    define USE_ASYNC_LDS 1
#  endif
#endif

#ifdef USE_ASYNC_LDS
__device__ __forceinline__ void async_b128(const void* g, void* l) {
  // signature (from compiler diagnostic): (v4i* src_global, v4i* dst_lds, Ii, Ii)
  __builtin_amdgcn_global_load_async_to_lds_b128((v4i_*)g, (v4i_*)l, 0, 0);
}
__device__ __forceinline__ void wait_async0() {
#  if __has_builtin(__builtin_amdgcn_s_wait_asynccnt)
  __builtin_amdgcn_s_wait_asynccnt(0);
#  else
  asm volatile("s_wait_asynccnt 0x0" ::: "memory");
#  endif
}
#endif

// ---------------------------------------------------------------------------
// Weight repack kernels: f32 -> f16, TRANSPOSED to [N][K] for the WMMA GEMM
// ---------------------------------------------------------------------------
__global__ void k_prep_B1t(const float* __restrict__ wg1, const float* __restrict__ wf1,
                           const float* __restrict__ ws1, f16* __restrict__ B1t) {
  int i = blockIdx.x * 256 + threadIdx.x;      // i = c*768 + r  (c: N idx, r: K idx)
  int c = i / 768, r = i % 768;
  int m = c / 256, co = c % 256;               // {gate|filter|skip} x out-channel
  int k = r / 256, ci = r % 256;               // tap k (t=7,9,11) x in-channel
  const float* w = (m == 0) ? wg1 : (m == 1) ? wf1 : ws1;  // (co, ci, k)
  B1t[i] = (f16)w[co * 768 + ci * 3 + k];
}

__global__ void k_prep_B2t(const float* sWq, const float* sWk, const float* sWv,
                           const float* dWq, const float* dWk, const float* dWv,
                           f16* __restrict__ B2t) {
  int i = blockIdx.x * 256 + threadIdx.x;      // i = c*256 + r
  int c = i / 256, r = i % 256;
  int g = c / 256, j = c % 256;
  const float* W = (g == 0) ? sWq : (g == 1) ? sWk : (g == 2) ? sWv
                 : (g == 3) ? dWq : (g == 4) ? dWk : dWv;
  B2t[i] = (f16)W[r * 256 + j];
}

__global__ void k_cvt_T(const float* __restrict__ src, f16* __restrict__ dst) {
  int i = blockIdx.x * 256 + threadIdx.x;      // i = j*256 + r ; src is [r][j]
  int j = i / 256, r = i % 256;
  dst[i] = (f16)src[r * 256 + j];
}

// ---------------------------------------------------------------------------
// Front: residual (last timestep) + layer-0 gated conv + LN + ReLU at t=7,9,11
// ---------------------------------------------------------------------------
__global__ void k_front(const float* __restrict__ x,
                        const float* wg0, const float* bg0,
                        const float* wf0, const float* bf0,
                        const float* ws0, const float* bs0,
                        const float* ln0g, const float* ln0b,
                        const float* wsk, const float* bsk,
                        f16* __restrict__ y0h, float* __restrict__ resl) {
  int bn = blockIdx.x; int c = threadIdx.x;
  __shared__ float red[256];
  const float xl0 = x[(bn * TT + 11) * 2 + 0];
  const float xl1 = x[(bn * TT + 11) * 2 + 1];
  resl[(size_t)bn * 256 + c] = xl0 * wsk[c] + xl1 * wsk[256 + c] + bsk[c];
  for (int j = 0; j < 3; ++j) {
    int t0 = 7 + 2 * j;
    float vg = bg0[c], vf = bf0[c], vs = bs0[c];
#pragma unroll
    for (int k = 0; k < 3; ++k) {
      int t = t0 - 2 + k;                      // causal taps (dil=1)
      float a0 = x[(bn * TT + t) * 2 + 0];
      float a1 = x[(bn * TT + t) * 2 + 1];
      vg += wg0[c * 6 + k] * a0 + wg0[c * 6 + 3 + k] * a1;
      vf += wf0[c * 6 + k] * a0 + wf0[c * 6 + 3 + k] * a1;
      vs += ws0[c * 6 + k] * a0 + ws0[c * 6 + 3 + k] * a1;
    }
    float g = 1.f / (1.f + __expf(-vg));
    float y = g * vf + (1.f - g) * vs;
    red[c] = y; __syncthreads();
    for (int s = 128; s > 0; s >>= 1) { if (c < s) red[c] += red[c + s]; __syncthreads(); }
    float mean = red[0] * (1.f / 256.f); __syncthreads();
    float d = y - mean; red[c] = d * d; __syncthreads();
    for (int s = 128; s > 0; s >>= 1) { if (c < s) red[c] += red[c + s]; __syncthreads(); }
    float var = red[0] * (1.f / 256.f); __syncthreads();
    float yn = (y - mean) * rsqrtf(var + 1e-5f) * ln0g[c] + ln0b[c];
    yn = yn > 0.f ? yn : 0.f;
    y0h[(size_t)bn * 768 + j * 256 + c] = (f16)yn;
  }
}

// ---------------------------------------------------------------------------
// WMMA GEMM: C[M,N] = A[M,K] * Bt[N,K]^T   (A,Bt f16 row-major, C f32)
// Requirements (guaranteed by caller via padding):
//   M % 64 == 0, Kd % 32 == 0, Bt has >= ceil128(N-tile) rows per batch,
//   all operand rows 16B-aligned.  Only C stores are guarded (n < N).
// Block: 256 threads / 8 waves; tile 64(M) x 128(N); K-chunk 32.
// Global -> LDS staging uses GLOBAL_LOAD_ASYNC_TO_LDS_B128 when available.
// ---------------------------------------------------------------------------
__global__ void k_gemm_wmma(const f16* __restrict__ A, const f16* __restrict__ Bt,
                            float* __restrict__ C, int M, int N, int Kd,
                            long sA, long sB, long sC) {
  A  += (long)blockIdx.z * sA;
  Bt += (long)blockIdx.z * sB;
  C  += (long)blockIdx.z * sC;
  constexpr int LDA = 40, LDB = 40;            // halves; 80B rows (16B aligned)
  __shared__ __align__(16) f16 As[64 * LDA];
  __shared__ __align__(16) f16 Bs[128 * LDB];
  const int tid = threadIdx.x, lane = tid & 31, wid = tid >> 5;
  const int mi  = wid & 3;                     // M subtile 0..3
  const int nio = (wid >> 2) * 4;              // N subtiles nio..nio+3
  const long mBlock = (long)blockIdx.y * 64;
  const long nBlock = (long)blockIdx.x * 128;
  v8f acc0 = {}, acc1 = {}, acc2 = {}, acc3 = {};
  const int arow = tid >> 2, acol = (tid & 3) * 8;   // A fill: 64x32, 8 h/thread
  const int brow = tid >> 1, bcol = (tid & 1) * 16;  // B fill: 128x32, 16 h/thread
  const f16* Aptr = A  + (mBlock + arow) * (long)Kd + acol;
  const f16* Bptr = Bt + (nBlock + brow) * (long)Kd + bcol;
  const int frow = mi * 16 + (lane & 15);      // A fragment row
  const int half = lane >> 4;
  const int nChunks = Kd >> 5;
  for (int kc = 0; kc < nChunks; ++kc) {
    const int kb = kc << 5;
    if (kc + 1 < nChunks) {                    // gfx1250 global_prefetch_b8
      __builtin_prefetch(Aptr + kb + 32, 0, 1);
      __builtin_prefetch(Bptr + kb + 32, 0, 1);
    }
#ifdef USE_ASYNC_LDS
    // Async DMA: one instruction per 16B, tracked by ASYNCcnt (no VGPR bounce)
    async_b128(Aptr + kb,     &As[arow * LDA + acol]);
    async_b128(Bptr + kb,     &Bs[brow * LDB + bcol]);
    async_b128(Bptr + kb + 8, &Bs[brow * LDB + bcol + 8]);
    wait_async0();
#else
    *(v8h*)&As[arow * LDA + acol]     = *(const v8h*)(Aptr + kb);
    *(v8h*)&Bs[brow * LDB + bcol]     = *(const v8h*)(Bptr + kb);
    *(v8h*)&Bs[brow * LDB + bcol + 8] = *(const v8h*)(Bptr + kb + 8);
#endif
    __syncthreads();
    // A fragment 16x32: lanes 0-15 hold K 0-7 & 16-23, lanes 16-31 K 8-15 & 24-31
    v8h alo = *(const v8h*)&As[frow * LDA + half * 8];
    v8h ahi = *(const v8h*)&As[frow * LDA + 16 + half * 8];
    v16h af = __builtin_shufflevector(alo, ahi, 0,1,2,3,4,5,6,7,8,9,10,11,12,13,14,15);
#pragma unroll
    for (int u = 0; u < 4; ++u) {
      int col = (nio + u) * 16 + (lane & 15);
      v8h blo = *(const v8h*)&Bs[col * LDB + half * 8];
      v8h bhi = *(const v8h*)&Bs[col * LDB + 16 + half * 8];
      v16h bf = __builtin_shufflevector(blo, bhi, 0,1,2,3,4,5,6,7,8,9,10,11,12,13,14,15);
      v8f* a = (u == 0) ? &acc0 : (u == 1) ? &acc1 : (u == 2) ? &acc2 : &acc3;
      *a = __builtin_amdgcn_wmma_f32_16x16x32_f16(false, af, false, bf, (short)0, *a, false, false);
    }
    __syncthreads();
  }
  // C store: 16x16 f32 layout (lanes 0-15 -> M=i, lanes 16-31 -> M=8+i)
#pragma unroll
  for (int u = 0; u < 4; ++u) {
    const v8f* a = (u == 0) ? &acc0 : (u == 1) ? &acc1 : (u == 2) ? &acc2 : &acc3;
    long n = nBlock + (nio + u) * 16 + (lane & 15);
    if (n < N) {
#pragma unroll
      for (int i2 = 0; i2 < 8; ++i2) {
        long gm = mBlock + mi * 16 + i2 + 8 * half;
        C[gm * N + n] = (*a)[i2];
      }
    }
  }
}

// ---------------------------------------------------------------------------
// Layer-1 epilogue: gate mix + LN + ReLU + residual -> hl (f16)
// ---------------------------------------------------------------------------
__global__ void k_epi1(const float* __restrict__ G1, const float* __restrict__ resl,
                       const float* bg1, const float* bf1c, const float* bs1,
                       const float* ln1g, const float* ln1b, f16* __restrict__ hlh) {
  int row = blockIdx.x; int c = threadIdx.x;
  __shared__ float red[256];
  float g = 1.f / (1.f + __expf(-(G1[(size_t)row * 768 + c] + bg1[c])));
  float f = G1[(size_t)row * 768 + 256 + c] + bf1c[c];
  float s = G1[(size_t)row * 768 + 512 + c] + bs1[c];
  float y = g * f + (1.f - g) * s;
  red[c] = y; __syncthreads();
  for (int st = 128; st > 0; st >>= 1) { if (c < st) red[c] += red[c + st]; __syncthreads(); }
  float mean = red[0] * (1.f / 256.f); __syncthreads();
  float d = y - mean; red[c] = d * d; __syncthreads();
  for (int st = 128; st > 0; st >>= 1) { if (c < st) red[c] += red[c + st]; __syncthreads(); }
  float var = red[0] * (1.f / 256.f); __syncthreads();
  float yn = (y - mean) * rsqrtf(var + 1e-5f) * ln1g[c] + ln1b[c];
  yn = yn > 0.f ? yn : 0.f;
  hlh[(size_t)row * 256 + c] = (f16)(yn + resl[(size_t)row * 256 + c]);
}

// ---------------------------------------------------------------------------
// Fixed-graph GAT: 9 structured edges per node, softmax per (node, head)
// QKV groups: sq=0, sk=256, sv=512, dq=768, dk=1024, dv=1280
// ---------------------------------------------------------------------------
__global__ void k_fixed_gat(const float* __restrict__ QKV, const float* __restrict__ edge_feat,
                            const int* __restrict__ eidx,
                            const float* sWe, const float* saq, const float* sak, const float* sae,
                            float* __restrict__ hf, f16* __restrict__ hfh) {
  int b = blockIdx.x / NN, n = blockIdx.x % NN;
  int tid = threadIdx.x;
  __shared__ float sS[9][8];
  __shared__ float sAl[9][8];
  if (tid < 72) {
    int o = tid / 8, h = tid % 8;
    int l = n * 9 + o;
    int snd = eidx[LL + l];
    const float* qrow = QKV + (size_t)(b * NN + n)   * 1536 + 0   + h * 32;
    const float* krow = QKV + (size_t)(b * NN + snd) * 1536 + 256 + h * 32;
    float ef0 = edge_feat[((size_t)b * LL + l) * 2 + 0];
    float ef1 = edge_feat[((size_t)b * LL + l) * 2 + 1];
    float s = 0.f;
    for (int d = 0; d < 32; ++d) {
      s += qrow[d] * saq[h * 32 + d];
      s += krow[d] * sak[h * 32 + d];
      float e = ef0 * sWe[h * 32 + d] + ef1 * sWe[256 + h * 32 + d];
      s += e * sae[h * 32 + d];
    }
    sS[o][h] = (s < 0.f) ? 0.2f * s : s;
  }
  __syncthreads();
  if (tid < 8) {
    int h = tid;
    float mx = -1e30f;
    for (int o = 0; o < 9; ++o) mx = fmaxf(mx, sS[o][h]);
    float den = 0.f;
    for (int o = 0; o < 9; ++o) den += __expf(sS[o][h] - mx);
    den = fmaxf(den, 1e-12f);
    for (int o = 0; o < 9; ++o) sAl[o][h] = __expf(sS[o][h] - mx) / den;
  }
  __syncthreads();
  {
    int h = tid >> 5, d = tid & 31;
    float acc = 0.f;
    for (int o = 0; o < 9; ++o) {
      int snd = eidx[LL + n * 9 + o];
      acc += sAl[o][h] * QKV[(size_t)(b * NN + snd) * 1536 + 512 + h * 32 + d];
    }
    float e = acc > 0.f ? acc : (__expf(acc) - 1.f);
    hf[(size_t)(b * NN + n) * 256 + tid] = e;
    hfh[(size_t)(b * NN + n) * 256 + tid] = (f16)e;
  }
}

// ---------------------------------------------------------------------------
// Adaptive adjacency: softmax(relu(E1 E2^T)) + top-16 threshold per row
// ---------------------------------------------------------------------------
__global__ void k_adj(const float* __restrict__ E1, const float* __restrict__ E2,
                      float* __restrict__ adjw, float* __restrict__ logA) {
  int n = blockIdx.x; int tid = threadIdx.x;  // 128 threads
  __shared__ float av[NN];
  __shared__ float wk[NN];
  __shared__ float red[128];
  __shared__ float kth;
  for (int j = tid; j < NN; j += 128) {
    float s = 0.f;
    for (int e = 0; e < 10; ++e) s += E1[n * 10 + e] * E2[j * 10 + e];
    av[j] = s > 0.f ? s : 0.f;
  }
  __syncthreads();
  float lm = -1e30f;
  for (int j = tid; j < NN; j += 128) lm = fmaxf(lm, av[j]);
  red[tid] = lm; __syncthreads();
  for (int s = 64; s > 0; s >>= 1) { if (tid < s) red[tid] = fmaxf(red[tid], red[tid + s]); __syncthreads(); }
  float mx = red[0]; __syncthreads();
  float ls = 0.f;
  for (int j = tid; j < NN; j += 128) ls += __expf(av[j] - mx);
  red[tid] = ls; __syncthreads();
  for (int s = 64; s > 0; s >>= 1) { if (tid < s) red[tid] += red[tid + s]; __syncthreads(); }
  float den = red[0]; __syncthreads();
  for (int j = tid; j < NN; j += 128) { av[j] = __expf(av[j] - mx) / den; wk[j] = av[j]; }
  __syncthreads();
  if (tid == 0) {
    float last = 0.f;
    for (int it = 0; it < 16; ++it) {
      float best = -1e30f; int bi = 0;
      for (int j = 0; j < NN; ++j) if (wk[j] > best) { best = wk[j]; bi = j; }
      wk[bi] = -1e30f; last = best;
    }
    kth = last;
  }
  __syncthreads();
  for (int j = tid; j < NN; j += 128) {
    float a = (av[j] >= kth) ? av[j] : 0.f;
    adjw[n * NN + j] = a;
    logA[n * NN + j] = __logf(fmaxf(a, 1e-12f));
  }
}

// ---------------------------------------------------------------------------
// Adaptive attention scores
// ---------------------------------------------------------------------------
__global__ void k_aqak(const float* __restrict__ QKV, const float* daq, const float* dak,
                       float* __restrict__ aq, float* __restrict__ ak) {
  int i = blockIdx.x * 256 + threadIdx.x;      // (b*400+n)*8 + h
  int h = i & 7; int row = i >> 3;
  const float* q = QKV + (size_t)row * 1536 + 768  + h * 32;
  const float* k = QKV + (size_t)row * 1536 + 1024 + h * 32;
  float sq = 0.f, sk = 0.f;
  for (int d = 0; d < 32; ++d) { sq += q[d] * daq[h * 32 + d]; sk += k[d] * dak[h * 32 + d]; }
  aq[i] = sq; ak[i] = sk;
}

// Vdt: transposed & padded V for batched GEMM: [b*8+h][d(128 pad)][j(416 pad)]
__global__ void k_vdt(const float* __restrict__ QKV, f16* __restrict__ Vdt) {
  int i = blockIdx.x * 256 + threadIdx.x;      // ((bh)*128 + d)*416 + j
  int j = i % KP; int r = i / KP; int d = r % NPB; int bh = r / NPB;
  int h = bh & 7; int b = bh >> 3;
  f16 v = (f16)0.f;
  if (d < 32 && j < NN)
    v = (f16)QKV[(size_t)(b * NN + j) * 1536 + 1280 + h * 32 + d];
  Vdt[i] = v;
}

__global__ void k_mden(const float* __restrict__ aq, const float* __restrict__ ak,
                       const float* __restrict__ adjw, const float* __restrict__ logA,
                       float* __restrict__ mb, float* __restrict__ db) {
  int i = blockIdx.x * 256 + threadIdx.x;      // (b*400+n)*8 + h
  int h = i & 7; int r = i >> 3; int n = r % NN; int b = r / NN;
  float aqi = aq[i];
  float M = -1e30f, S = 0.f;
  for (int j = 0; j < NN; ++j) {
    float s;
    if (adjw[n * NN + j] > 0.f) {
      float t = aqi + ak[(b * NN + j) * 8 + h];
      t = t < 0.f ? 0.2f * t : t;
      s = t + logA[n * NN + j];
    } else s = -1e9f;
    if (s > M) { S = S * __expf(M - s) + 1.f; M = s; }
    else       { S += __expf(s - M); }
  }
  mb[i] = M; db[i] = fmaxf(S, 1e-30f);
}

// AL: padded attention matrix per (b,h): [448 rows][416 cols] f16, zero pad
__global__ void k_al(const float* __restrict__ aq, const float* __restrict__ ak,
                     const float* __restrict__ adjw, const float* __restrict__ logA,
                     const float* __restrict__ mb, const float* __restrict__ db,
                     f16* __restrict__ AL) {
  int i = blockIdx.x * 256 + threadIdx.x;      // ((bh)*448 + n)*416 + j
  int j = i % KP; int r = i / KP; int n = r % MP; int bh = r / MP;
  int h = bh & 7; int b = bh >> 3;
  f16 out = (f16)0.f;
  if (n < NN && j < NN) {
    float s;
    if (adjw[n * NN + j] > 0.f) {
      float t = aq[(b * NN + n) * 8 + h] + ak[(b * NN + j) * 8 + h];
      t = t < 0.f ? 0.2f * t : t;
      s = t + logA[n * NN + j];
    } else s = -1e9f;
    int mi = (b * NN + n) * 8 + h;
    out = (f16)(__expf(s - mb[mi]) / db[mi]);
  }
  AL[i] = out;
}

__global__ void k_merge(const float* __restrict__ agg, const float* __restrict__ hf,
                        f16* __restrict__ hah, f16* __restrict__ hsum) {
  int i = blockIdx.x * 256 + threadIdx.x;      // row*256 + (h*32+d)
  int c = i & 255; int row = i >> 8; int n = row % NN; int b = row / NN;
  int h = c >> 5, d = c & 31;
  float v = agg[(size_t)((b * 8 + h) * MP + n) * 32 + d];
  float e = v > 0.f ? v : (__expf(v) - 1.f);
  hah[i] = (f16)e;
  hsum[i] = (f16)(e + hf[i]);
}

// ---------------------------------------------------------------------------
// Final fusion + output projection (HID x 3)
// ---------------------------------------------------------------------------
__global__ void k_final(const float* __restrict__ U1, const float* __restrict__ U2,
                        const float* __restrict__ U3,
                        const float* fb1, const float* fb2, const float* fb3,
                        const float* Wo, const float* bo, float* __restrict__ out) {
  int row = blockIdx.x; int c = threadIdx.x;
  __shared__ float fu[256];
  float g = 1.f / (1.f + __expf(-(U1[(size_t)row * 256 + c] + fb1[c])));
  float a = U2[(size_t)row * 256 + c] + fb2[c];
  float s = U3[(size_t)row * 256 + c] + fb3[c];
  fu[c] = tanhf(g * a + (1.f - g) * s);
  __syncthreads();
  if (c < 3) {
    float acc = bo[c];
    for (int k = 0; k < 256; ++k) acc += fu[k] * Wo[k * 3 + c];
    out[row * 3 + c] = acc;
  }
}

// ---------------------------------------------------------------------------
extern "C" void kernel_launch(void* const* d_in, const int* in_sizes, int n_in,
                              void* d_out, int out_size, void* d_ws, size_t ws_size,
                              hipStream_t stream) {
  const float* x    = (const float*)d_in[0];
  const float* edgf = (const float*)d_in[1];
  const int*   eidx = (const int*)d_in[2];
  const float* wg0  = (const float*)d_in[3];  const float* bg0 = (const float*)d_in[4];
  const float* wf0  = (const float*)d_in[5];  const float* bf0 = (const float*)d_in[6];
  const float* ws0  = (const float*)d_in[7];  const float* bs0 = (const float*)d_in[8];
  const float* ln0g = (const float*)d_in[9];  const float* ln0b = (const float*)d_in[10];
  const float* wg1  = (const float*)d_in[11]; const float* bg1 = (const float*)d_in[12];
  const float* wf1  = (const float*)d_in[13]; const float* bf1 = (const float*)d_in[14];
  const float* ws1  = (const float*)d_in[15]; const float* bs1 = (const float*)d_in[16];
  const float* ln1g = (const float*)d_in[17]; const float* ln1b = (const float*)d_in[18];
  const float* wsk  = (const float*)d_in[19]; const float* bsk = (const float*)d_in[20];
  const float* sWq  = (const float*)d_in[21]; const float* sWk = (const float*)d_in[22];
  const float* sWv  = (const float*)d_in[23]; const float* sWe = (const float*)d_in[24];
  const float* saq  = (const float*)d_in[25]; const float* sak = (const float*)d_in[26];
  const float* sae  = (const float*)d_in[27];
  const float* dWq  = (const float*)d_in[28]; const float* dWk = (const float*)d_in[29];
  const float* dWv  = (const float*)d_in[30];
  const float* daq  = (const float*)d_in[31]; const float* dak = (const float*)d_in[32];
  const float* E1   = (const float*)d_in[33]; const float* E2  = (const float*)d_in[34];
  const float* fW1  = (const float*)d_in[35]; const float* fb1 = (const float*)d_in[36];
  const float* fW2  = (const float*)d_in[37]; const float* fb2 = (const float*)d_in[38];
  const float* fW3  = (const float*)d_in[39]; const float* fb3 = (const float*)d_in[40];
  const float* Wo   = (const float*)d_in[41]; const float* bo  = (const float*)d_in[42];
  (void)in_sizes; (void)n_in; (void)out_size; (void)ws_size;

  char* wsb = (char*)d_ws;
  size_t off = 0;
  auto alloc = [&](size_t bytes) -> void* {
    void* p = wsb + off; off += (bytes + 255) & ~(size_t)255; return p;
  };
  f16*   B1t  = (f16*)  alloc((size_t)768 * 768 * 2);
  f16*   B2t  = (f16*)  alloc((size_t)1536 * 256 * 2);
  f16*   Bf1t = (f16*)  alloc((size_t)65536 * 2);
  f16*   Bf2t = (f16*)  alloc((size_t)65536 * 2);
  f16*   Bf3t = (f16*)  alloc((size_t)65536 * 2);
  f16*   y0h  = (f16*)  alloc((size_t)BN * 768 * 2);
  float* resl = (float*)alloc((size_t)BN * 256 * 4);
  float* G1   = (float*)alloc((size_t)BN * 768 * 4);
  f16*   hlh  = (f16*)  alloc((size_t)BN * 256 * 2);
  float* QKV  = (float*)alloc((size_t)BN * 1536 * 4);
  float* hfb  = (float*)alloc((size_t)BN * 256 * 4);
  f16*   hfh  = (f16*)  alloc((size_t)BN * 256 * 2);
  float* adjw = (float*)alloc((size_t)NN * NN * 4);
  float* logA = (float*)alloc((size_t)NN * NN * 4);
  float* aqb  = (float*)alloc((size_t)51200 * 4);
  float* akb  = (float*)alloc((size_t)51200 * 4);
  float* mbuf = (float*)alloc((size_t)51200 * 4);
  float* dbuf = (float*)alloc((size_t)51200 * 4);
  f16*   AL   = (f16*)  alloc((size_t)128 * MP * KP * 2);   // 128 x 448 x 416
  f16*   Vdt  = (f16*)  alloc((size_t)128 * NPB * KP * 2);  // 128 x 128 x 416
  float* aggb = (float*)alloc((size_t)128 * MP * 32 * 4);   // 128 x 448 x 32
  f16*   hah  = (f16*)  alloc((size_t)BN * 256 * 2);
  f16*   hsum = (f16*)  alloc((size_t)BN * 256 * 2);
  float* U1   = (float*)alloc((size_t)BN * 256 * 4);
  float* U2   = (float*)alloc((size_t)BN * 256 * 4);
  float* U3   = (float*)alloc((size_t)BN * 256 * 4);

  // Weight repack (transposed to [N][K])
  k_prep_B1t<<<2304, 256, 0, stream>>>(wg1, wf1, ws1, B1t);
  k_prep_B2t<<<1536, 256, 0, stream>>>(sWq, sWk, sWv, dWq, dWk, dWv, B2t);
  k_cvt_T<<<256, 256, 0, stream>>>(fW1, Bf1t);
  k_cvt_T<<<256, 256, 0, stream>>>(fW2, Bf2t);
  k_cvt_T<<<256, 256, 0, stream>>>(fW3, Bf3t);

  // Front (layer-0 + residual, last-timestep-only)
  k_front<<<BN, 256, 0, stream>>>(x, wg0, bg0, wf0, bf0, ws0, bs0,
                                  ln0g, ln0b, wsk, bsk, y0h, resl);

  // Layer-1 as one WMMA GEMM: (6400 x 768) x (768 x 768)
  k_gemm_wmma<<<dim3(6, 100, 1), 256, 0, stream>>>(y0h, B1t, G1, BN, 768, 768, 0, 0, 0);
  k_epi1<<<BN, 256, 0, stream>>>(G1, resl, bg1, bf1, bs1, ln1g, ln1b, hlh);

  // All six projections as one WMMA GEMM: (6400 x 256) x (256 x 1536)
  k_gemm_wmma<<<dim3(12, 100, 1), 256, 0, stream>>>(hlh, B2t, QKV, BN, 1536, 256, 0, 0, 0);

  // Fixed-graph attention
  k_fixed_gat<<<BN, 256, 0, stream>>>(QKV, edgf, eidx, sWe, saq, sak, sae, hfb, hfh);

  // Adaptive adjacency + attention
  k_adj<<<NN, 128, 0, stream>>>(E1, E2, adjw, logA);
  k_aqak<<<200, 256, 0, stream>>>(QKV, daq, dak, aqb, akb);
  k_vdt<<<(128 * NPB * KP) / 256, 256, 0, stream>>>(QKV, Vdt);
  k_mden<<<200, 256, 0, stream>>>(aqb, akb, adjw, logA, mbuf, dbuf);
  k_al<<<(128 * MP * KP) / 256, 256, 0, stream>>>(aqb, akb, adjw, logA, mbuf, dbuf, AL);

  // Batched attention aggregation: 128 x [(448x416) x (416x32)] via WMMA
  k_gemm_wmma<<<dim3(1, MP / 64, 128), 256, 0, stream>>>(AL, Vdt, aggb, MP, 32, KP,
                                                         (long)MP * KP, (long)NPB * KP,
                                                         (long)MP * 32);
  k_merge<<<6400, 256, 0, stream>>>(aggb, hfb, hah, hsum);

  // Fusion GEMMs: (6400 x 256) x (256 x 256)
  k_gemm_wmma<<<dim3(2, 100, 1), 256, 0, stream>>>(hsum, Bf1t, U1, BN, 256, 256, 0, 0, 0);
  k_gemm_wmma<<<dim3(2, 100, 1), 256, 0, stream>>>(hfh,  Bf2t, U2, BN, 256, 256, 0, 0, 0);
  k_gemm_wmma<<<dim3(2, 100, 1), 256, 0, stream>>>(hah,  Bf3t, U3, BN, 256, 256, 0, 0, 0);

  k_final<<<BN, 256, 0, stream>>>(U1, U2, U3, fb1, fb2, fb3, Wo, bo, (float*)d_out);
}
